// SpatialAttention_86397562126360
// MI455X (gfx1250) — matrix operand order repaired and checked by reading, not compiled, not run on hardware
//
#include <hip/hip_runtime.h>

// ---------------- types / helpers ----------------
typedef __bf16  bf16x16 __attribute__((ext_vector_type(16)));
typedef float   f32x8   __attribute__((ext_vector_type(8)));
typedef int     v4i     __attribute__((ext_vector_type(4)));
typedef unsigned short u16;
typedef unsigned int   u32;

union Frag16 {
  uint4   u[2];
  u16     s[16];
  bf16x16 v;
};

__device__ __forceinline__ u16 f2bf(float f) {
  union { float f; u32 u; } c; c.f = f;
  u32 r = c.u + 0x7fffu + ((c.u >> 16) & 1u);   // round-to-nearest-even
  return (u16)(r >> 16);
}
__device__ __forceinline__ float bf2f(u16 h) {
  union { u32 u; float f; } c; c.u = ((u32)h) << 16;
  return c.f;
}
__device__ __forceinline__ f32x8 wmma_bf16(const Frag16& a, const Frag16& b, f32x8 c) {
  return __builtin_amdgcn_wmma_f32_16x16x32_bf16(false, a.v, false, b.v,
                                                 (short)0, c, false, false);
}

// ---------------- gfx1250 async global->LDS path (guarded) ----------------
#if defined(__AMDGCN__) && __has_builtin(__builtin_amdgcn_global_load_async_to_lds_b128)
#define HAS_ASYNC_LDS 1
#else
#define HAS_ASYNC_LDS 0
#endif

// copy 16 bytes global -> LDS (async on gfx1250, sync fallback otherwise)
__device__ __forceinline__ void async_copy16(const void* gsrc, void* ldst) {
#if HAS_ASYNC_LDS
  __builtin_amdgcn_global_load_async_to_lds_b128(
      (v4i*)gsrc,
      (__attribute__((address_space(3))) v4i*)ldst,
      0, 0);
#else
  *(uint4*)ldst = *(const uint4*)gsrc;
#endif
}

template <int N>
__device__ __forceinline__ void wait_asynccnt() {
#if HAS_ASYNC_LDS
#if __has_builtin(__builtin_amdgcn_s_wait_asynccnt)
  __builtin_amdgcn_s_wait_asynccnt(N);
#else
  asm volatile("s_wait_asynccnt %0" ::"i"(N) : "memory");
#endif
#endif
}

// ---------------- constants ----------------
#define S_LEN  1024
#define T_LEN  8
#define HID    1152
#define NHEAD  16
#define HD     72
#define MROWS  (T_LEN * S_LEN)   // 8192

// ---------------- fp32 -> bf16 convert (linear) ----------------
__global__ void cvt_f32_bf16(const float* __restrict__ src, u16* __restrict__ dst, int n) {
  int i = blockIdx.x * blockDim.x + threadIdx.x;
  if (i < n) dst[i] = f2bf(src[i]);
}

// ---------------- fp32 [K][N] -> bf16 transposed [N][K] (tiled) ----------------
__global__ __launch_bounds__(256) void cvt_transpose_bf16(
    const float* __restrict__ src, u16* __restrict__ dst, int K, int N) {
  __shared__ float tile[32][33];
  const int n0 = blockIdx.x * 32;
  const int k0 = blockIdx.y * 32;
  const int tx = threadIdx.x & 31;
  const int ty = threadIdx.x >> 5;          // 0..7
#pragma unroll
  for (int i = 0; i < 32; i += 8)
    tile[ty + i][tx] = src[(size_t)(k0 + ty + i) * N + n0 + tx];
  __syncthreads();
#pragma unroll
  for (int i = 0; i < 32; i += 8)
    dst[(size_t)(n0 + ty + i) * K + k0 + tx] = f2bf(tile[tx][ty + i]);
}

// ---------------- tiled bf16 GEMM: C[M,N] = A[M,K] * BT[N,K]^T + bias ----------------
// Block: 256 threads (8 waves). Tile: BM=128, BN=64, BK=32. Double-buffered LDS,
// tiles staged with async global->LDS B128 copies (ASYNCcnt) overlapping WMMA compute.
template <bool OUT_F32>
__global__ __launch_bounds__(256) void gemm_bf16(
    const u16* __restrict__ A, const u16* __restrict__ BT,
    const float* __restrict__ bias, void* __restrict__ Cout,
    int K, int lda, int ldb, int ldc)
{
  __shared__ __align__(16) u16 Alds[2][128 * 32];   // [m][k]
  __shared__ __align__(16) u16 Blds[2][64 * 32];    // [n][k]

  const int n0   = blockIdx.x * 64;
  const int m0   = blockIdx.y * 128;
  const int tid  = threadIdx.x;
  const int lane = tid & 31;
  const int wave = tid >> 5;
  const int half = lane >> 4;
  const int l16  = lane & 15;

  f32x8 acc[4] = {};

  // stage tile kt into buffer buf: 3 x 16B async copies per thread
  auto stage = [&](int kt, int buf) {
#pragma unroll
    for (int i = 0; i < 2; ++i) {
      const int idx = tid + i * 256;
      const int r = idx >> 2, c8 = (idx & 3) * 8;
      async_copy16(&A[(size_t)(m0 + r) * lda + kt + c8], &Alds[buf][r * 32 + c8]);
    }
    {
      const int r = tid >> 2, c8 = (tid & 3) * 8;
      async_copy16(&BT[(size_t)(n0 + r) * ldb + kt + c8], &Blds[buf][r * 32 + c8]);
    }
  };

  stage(0, 0);
  int cur = 0;
  for (int kt = 0; kt < K; kt += 32, cur ^= 1) {
    __syncthreads();                    // everyone done reading buffer cur^1
    const bool more = (kt + 32) < K;
    if (more) {
      stage(kt + 32, cur ^ 1);          // prefetch next tile (in flight during compute)
      wait_asynccnt<3>();               // current tile's 3 copies have landed
    } else {
      wait_asynccnt<0>();
    }
    __syncthreads();                    // all threads' current tile visible

    Frag16 a;
    const int arow = (wave << 4) + l16;
    a.u[0] = *(const uint4*)&Alds[cur][arow * 32 + half * 8];
    a.u[1] = *(const uint4*)&Alds[cur][arow * 32 + half * 8 + 16];
#pragma unroll
    for (int j = 0; j < 4; ++j) {
      Frag16 b;
      const int bn  = j * 16 + l16;
      const int kk0 = half * 8;
      b.u[0] = *(const uint4*)&Blds[cur][bn * 32 + kk0];
      b.u[1] = *(const uint4*)&Blds[cur][bn * 32 + kk0 + 16];
      acc[j] = wmma_bf16(a, b, acc[j]);
    }
  }

  // epilogue: C layout -> global
#pragma unroll
  for (int j = 0; j < 4; ++j) {
    const int n = n0 + j * 16 + l16;
    const float bv = bias ? bias[n] : 0.0f;
#pragma unroll
    for (int r = 0; r < 8; ++r) {
      const int m = m0 + wave * 16 + r + 8 * half;
      const float v = acc[j][r] + bv;
      if (OUT_F32) ((float*)Cout)[(size_t)m * ldc + n] = v;
      else         ((u16*)Cout)[(size_t)m * ldc + n]   = f2bf(v);
    }
  }
}

// ---------------- per-(row,head) RMSNorm over D=72, in place on bf16 ----------------
__global__ __launch_bounds__(256) void rmsnorm_bf16(
    u16* __restrict__ buf, const float* __restrict__ gamma, int ld)
{
  const int lane = threadIdx.x & 31;
  const int wave = threadIdx.x >> 5;
  const int idx  = blockIdx.x * 8 + wave;           // row-head id
  const int row  = idx >> 4;
  const int h    = idx & 15;
  u16* p = buf + (size_t)row * ld + h * HD;

  float x0 = bf2f(p[lane]);
  float x1 = bf2f(p[lane + 32]);
  float x2 = (lane < 8) ? bf2f(p[lane + 64]) : 0.0f;
  float ss = x0 * x0 + x1 * x1 + x2 * x2;
#pragma unroll
  for (int d = 16; d >= 1; d >>= 1) ss += __shfl_xor(ss, d, 32);
  const float inv = rsqrtf(ss / (float)HD + 1e-6f);

  const float* g = gamma + h * HD;
  p[lane]      = f2bf(x0 * inv * g[lane]);
  p[lane + 32] = f2bf(x1 * inv * g[lane + 32]);
  if (lane < 8) p[lane + 64] = f2bf(x2 * inv * g[lane + 64]);
}

// ---------------- flash-style attention ----------------
// grid.x = 8 (q tiles of 128), grid.y = T*H = 128. Block = 256 threads (8 waves).
// Each wave owns 16 query rows. Keys streamed in chunks of 64. D padded 72->96 (scores)
// and 72->80 (output). K chunk staged with async B128 copies (72 bf16 = 9x16B per key).
__global__ __launch_bounds__(256) void attn_kernel(
    const u16* __restrict__ qb, const u16* __restrict__ kvb, u16* __restrict__ ob)
{
  __shared__ __align__(16) u16 Klds[64 * 96];        // [key][d]
  __shared__ __align__(16) u16 Vlds[80 * 64];        // transposed [n][key]
  __shared__ __align__(16) u16 Plds[8 * 16 * 64];    // per-wave P staging

  const int th   = blockIdx.y;
  const int t    = th >> 4;
  const int h    = th & 15;
  const int tid  = threadIdx.x;
  const int lane = tid & 31;
  const int wave = tid >> 5;
  const int half = lane >> 4;
  const int l16  = lane & 15;

  // zero K d-padding [72,96) once; never rewritten by the streaming copies
  for (int i = tid; i < 64 * 24; i += 256)
    Klds[(i / 24) * 96 + 72 + (i % 24)] = 0;

  // Q fragments held in registers: 3 k-chunks of 32 (d padded to 96)
  const int qrow = t * S_LEN + blockIdx.x * 128 + wave * 16 + l16;
  const u16* qp  = qb + (size_t)qrow * HID + h * HD;
  Frag16 qf[3];
#pragma unroll
  for (int ks = 0; ks < 3; ++ks) {
#pragma unroll
    for (int i = 0; i < 16; ++i) {
      const int d = ks * 32 + half * 8 + (i < 8 ? i : 16 + (i - 8));
      qf[ks].s[i] = (d < HD) ? qp[d] : (u16)0;
    }
  }

  f32x8 o[5] = {};
  float mrow[8], lrow[8];
#pragma unroll
  for (int r = 0; r < 8; ++r) { mrow[r] = -1e30f; lrow[r] = 0.0f; }

  for (int kb = 0; kb < S_LEN; kb += 64) {
    __syncthreads();
    // K chunk: 64 keys x 72 bf16 = 576 x 16B chunks, async to Klds[key][d]
#pragma unroll
    for (int i = 0; i < 3; ++i) {
      const int idx = tid + i * 256;
      if (idx < 576) {
        const int key = idx / 9, part = idx % 9;
        const u16* src =
            kvb + (size_t)(t * S_LEN + kb + key) * (2 * HID) + h * HD + part * 8;
        async_copy16(src, &Klds[key * 96 + part * 8]);
      }
    }
    // V chunk transposed into Vlds[n][key] (scalar scatter; zero-pads n>=72)
    {
      const int key  = tid >> 2;
      const int part = tid & 3;
      const u16* vp =
          kvb + (size_t)(t * S_LEN + kb + key) * (2 * HID) + HID + h * HD;
#pragma unroll
      for (int e = 0; e < 20; ++e) {
        const int n = part * 20 + e;
        Vlds[n * 64 + key] = (n < HD) ? vp[n] : (u16)0;
      }
    }
    wait_asynccnt<0>();
    __syncthreads();

    // S = Q @ K^T  (16 x 64 per wave)
    f32x8 s[4] = {};
#pragma unroll
    for (int j = 0; j < 4; ++j) {
      const int bn  = j * 16 + l16;
      const int kk0 = half * 8;
#pragma unroll
      for (int ks = 0; ks < 3; ++ks) {
        Frag16 b;
        b.u[0] = *(const uint4*)&Klds[bn * 96 + ks * 32 + kk0];
        b.u[1] = *(const uint4*)&Klds[bn * 96 + ks * 32 + kk0 + 16];
        s[j] = wmma_bf16(qf[ks], b, s[j]);
      }
    }

    // online softmax: stats per row (rows r+8*half), reduce across the 16-lane N group
    float mnew[8];
#pragma unroll
    for (int r = 0; r < 8; ++r) {
      mnew[r] = mrow[r];
#pragma unroll
      for (int j = 0; j < 4; ++j) mnew[r] = fmaxf(mnew[r], s[j][r]);
    }
#pragma unroll
    for (int d = 1; d < 16; d <<= 1)
#pragma unroll
      for (int r = 0; r < 8; ++r) mnew[r] = fmaxf(mnew[r], __shfl_xor(mnew[r], d, 32));

    float psum[8];
#pragma unroll
    for (int r = 0; r < 8; ++r) {
      const float sc = __expf(mrow[r] - mnew[r]);
      lrow[r] *= sc;
#pragma unroll
      for (int n = 0; n < 5; ++n) o[n][r] *= sc;
      psum[r] = 0.0f;
    }
#pragma unroll
    for (int j = 0; j < 4; ++j)
#pragma unroll
      for (int r = 0; r < 8; ++r) {
        const float p = __expf(s[j][r] - mnew[r]);
        s[j][r] = p;
        psum[r] += p;
      }
#pragma unroll
    for (int d = 1; d < 16; d <<= 1)
#pragma unroll
      for (int r = 0; r < 8; ++r) psum[r] += __shfl_xor(psum[r], d, 32);
#pragma unroll
    for (int r = 0; r < 8; ++r) { lrow[r] += psum[r]; mrow[r] = mnew[r]; }

    // P: C-layout -> LDS -> A-layout fragments
    u16* pl = &Plds[wave * 16 * 64];
#pragma unroll
    for (int j = 0; j < 4; ++j)
#pragma unroll
      for (int r = 0; r < 8; ++r)
        pl[(r + 8 * half) * 64 + j * 16 + l16] = f2bf(s[j][r]);
    __syncthreads();

    Frag16 pa[2];
#pragma unroll
    for (int kc = 0; kc < 2; ++kc) {
      pa[kc].u[0] = *(const uint4*)&pl[l16 * 64 + kc * 32 + half * 8];
      pa[kc].u[1] = *(const uint4*)&pl[l16 * 64 + kc * 32 + half * 8 + 16];
    }

    // O += P @ V  (16 x 80 per wave)
#pragma unroll
    for (int n = 0; n < 5; ++n) {
      const int bn  = n * 16 + l16;
      const int kk0 = half * 8;
#pragma unroll
      for (int kc = 0; kc < 2; ++kc) {
        Frag16 b;
        b.u[0] = *(const uint4*)&Vlds[bn * 64 + kc * 32 + kk0];
        b.u[1] = *(const uint4*)&Vlds[bn * 64 + kc * 32 + kk0 + 16];
        o[n] = wmma_bf16(pa[kc], b, o[n]);
      }
    }
  }

  // normalize + store bf16 attention output
  const int orow0 = t * S_LEN + blockIdx.x * 128 + wave * 16;
#pragma unroll
  for (int n = 0; n < 5; ++n) {
    const int col = n * 16 + l16;
    if (col < HD) {
#pragma unroll
      for (int r = 0; r < 8; ++r) {
        const int m = orow0 + r + 8 * half;
        ob[(size_t)m * HID + h * HD + col] = f2bf(o[n][r] / lrow[r]);
      }
    }
  }
}

// ---------------- launch ----------------
extern "C" void kernel_launch(void* const* d_in, const int* in_sizes, int n_in,
                              void* d_out, int out_size, void* d_ws, size_t ws_size,
                              hipStream_t stream) {
  (void)in_sizes; (void)n_in; (void)out_size; (void)ws_size;

  const float* x   = (const float*)d_in[0];
  const float* Wq  = (const float*)d_in[1];
  const float* bq  = (const float*)d_in[2];
  const float* Wkv = (const float*)d_in[3];
  const float* bkv = (const float*)d_in[4];
  const float* qg  = (const float*)d_in[5];
  const float* kg  = (const float*)d_in[6];
  const float* Wo  = (const float*)d_in[7];
  float* out = (float*)d_out;

  // workspace carve-out (bf16 buffers), all offsets 256B aligned
  char* ws = (char*)d_ws;
  u16* xb   = (u16*)ws; ws += (size_t)MROWS * HID * 2;        // 18.9 MB
  u16* WqT  = (u16*)ws; ws += (size_t)HID * HID * 2;          //  2.7 MB  [N][K]
  u16* WkvT = (u16*)ws; ws += (size_t)HID * 2 * HID * 2;      //  5.3 MB  [2N][K]
  u16* WoT  = (u16*)ws; ws += (size_t)HID * HID * 2;          //  2.7 MB  [N][K]
  u16* qbuf = (u16*)ws; ws += (size_t)MROWS * HID * 2;        // 18.9 MB
  u16* kvb  = (u16*)ws; ws += (size_t)MROWS * 2 * HID * 2;    // 37.7 MB
  u16* obuf = (u16*)ws; ws += (size_t)MROWS * HID * 2;        // 18.9 MB

  const int nx = MROWS * HID;
  cvt_f32_bf16<<<(nx + 255) / 256, 256, 0, stream>>>(x, xb, nx);
  // weights converted + transposed once (reused by every M-block)
  cvt_transpose_bf16<<<dim3(HID / 32, HID / 32), 256, 0, stream>>>(Wq,  WqT,  HID, HID);
  cvt_transpose_bf16<<<dim3(2 * HID / 32, HID / 32), 256, 0, stream>>>(Wkv, WkvT, HID, 2 * HID);
  cvt_transpose_bf16<<<dim3(HID / 32, HID / 32), 256, 0, stream>>>(Wo,  WoT,  HID, HID);

  // Q = x @ Wq + bq          (8192 x 1152)
  gemm_bf16<false><<<dim3(HID / 64, MROWS / 128), 256, 0, stream>>>(
      xb, WqT, bq, qbuf, HID, HID, HID, HID);
  // KV = x @ Wkv + bkv       (8192 x 2304); k = cols [0,1152), v = cols [1152,2304)
  gemm_bf16<false><<<dim3(2 * HID / 64, MROWS / 128), 256, 0, stream>>>(
      xb, WkvT, bkv, kvb, HID, HID, HID, 2 * HID);

  // QKNorm (per-head RMSNorm, in place)
  rmsnorm_bf16<<<MROWS * NHEAD / 8, 256, 0, stream>>>(qbuf, qg, HID);
  rmsnorm_bf16<<<MROWS * NHEAD / 8, 256, 0, stream>>>(kvb,  kg, 2 * HID);

  // attention (no 1/sqrt(d) scaling per reference)
  attn_kernel<<<dim3(S_LEN / 128, T_LEN * NHEAD), 256, 0, stream>>>(qbuf, kvb, obuf);

  // out = attn_out @ Wo  -> fp32 d_out
  gemm_bf16<true><<<dim3(HID / 64, MROWS / 128), 256, 0, stream>>>(
      obuf, WoT, nullptr, out, HID, HID, HID, HID);
}
